// SimpleRNN_73126113182042
// MI455X (gfx1250) — matrix operand order repaired and checked by reading, not compiled, not run on hardware
//
#include <hip/hip_runtime.h>
#include <hip/hip_bf16.h>

// ---------------------------------------------------------------------------
// 2-layer tanh RNN on gfx1250 via v_wmma_f32_16x16x32_f16.
//   ans   (S,B,V) f32  = y2 @ W_fc^T + b_fc
//   state (2,B,H) f32  = [h1_last, h2_last]
// Final projection stages its B-operand through LDS with async
// global->LDS copies (ASYNCcnt) shared by all 8 waves of a block.
// ---------------------------------------------------------------------------

static constexpr int HID  = 1024;  // HIDDEN
static constexpr int BAT  = 64;    // BATCH
static constexpr int SEQL = 128;   // SEQ
static constexpr int VOC  = 5000;  // VOCAB
static constexpr int VP   = 5056;  // VOCAB padded to multiple of 64 (316 n-tiles)

typedef __attribute__((ext_vector_type(8)))  _Float16 v8h;
typedef __attribute__((ext_vector_type(16))) _Float16 v16h;
typedef __attribute__((ext_vector_type(8)))  float    v8f;
typedef __attribute__((ext_vector_type(4)))  int      v4i;

// ---- WMMA fragment loaders (contiguous-K order, wave32) --------------------
// A 16x32 f16, row-major source (M,K), ld = K stride.
// lane L: m = L&15, half = L>>4 ; frag elem i <-> K = 16*half + i
__device__ __forceinline__ v16h load_a16x32(const _Float16* __restrict__ base, int ld) {
    const int lane = threadIdx.x & 31;
    const int m  = lane & 15;
    const int hh = lane >> 4;
    const _Float16* p = base + (size_t)m * ld + hh * 16;
    v8h lo = *reinterpret_cast<const v8h*>(p);        // K = 16h .. 16h+7
    v8h hi = *reinterpret_cast<const v8h*>(p + 8);    // K = 16h+8 .. 16h+15
    return __builtin_shufflevector(lo, hi, 0,1,2,3,4,5,6,7,8,9,10,11,12,13,14,15);
}

// B 32x16 f16 with B[k,n] = W[n,k], W row-major (N,K); wbase = &W[n0*ld].
// lane L: n = L&15, half = L>>4 ; frag elem i <-> K = 16*half + i (SWMMAC B table analog)
__device__ __forceinline__ v16h load_b32x16(const _Float16* __restrict__ wbase, int ld) {
    const int lane = threadIdx.x & 31;
    const int n  = lane & 15;
    const int hh = lane >> 4;
    const _Float16* p = wbase + (size_t)n * ld + hh * 16;
    v8h lo = *reinterpret_cast<const v8h*>(p);
    v8h hi = *reinterpret_cast<const v8h*>(p + 8);
    return __builtin_shufflevector(lo, hi, 0,1,2,3,4,5,6,7,8,9,10,11,12,13,14,15);
}

__device__ __forceinline__ v8f wmma_f16(v16h a, v16h b, v8f c) {
    return __builtin_amdgcn_wmma_f32_16x16x32_f16(false, a, false, b, (short)0, c, false, false);
}

// ---- async global -> LDS copy (gfx1250), with synchronous fallback ---------
#if defined(__has_builtin)
#if __has_builtin(__builtin_amdgcn_global_load_async_to_lds_b128)
#define HAVE_ASYNC_LDS 1
#endif
#endif

__device__ __forceinline__ void copy16_g2lds(const _Float16* gsrc, _Float16* ldst) {
#ifdef HAVE_ASYNC_LDS
    // builtin signature (from clang diagnostic): param0 = v4i __device__*  (as1),
    // param1 = LDS pointer, then imm offset + cpol.
    typedef __attribute__((address_space(1))) v4i g1_v4i;
    typedef __attribute__((address_space(3))) v4i l3_v4i;
    __builtin_amdgcn_global_load_async_to_lds_b128((g1_v4i*)gsrc, (l3_v4i*)ldst, 0, 0);
#else
    *reinterpret_cast<v8h*>(ldst) = *reinterpret_cast<const v8h*>(gsrc);
#endif
}

__device__ __forceinline__ void wait_async_copies() {
#ifdef HAVE_ASYNC_LDS
#if __has_builtin(__builtin_amdgcn_s_wait_asynccnt)
    __builtin_amdgcn_s_wait_asynccnt(0);
#else
    asm volatile("s_wait_asynccnt 0" ::: "memory");
#endif
#endif
}

// ---- prep kernels ----------------------------------------------------------
__global__ void k_cast_f16(const float* __restrict__ src, _Float16* __restrict__ dst, int n) {
    int i = blockIdx.x * blockDim.x + threadIdx.x;
    if (i < n) dst[i] = (_Float16)src[i];
}

// W_fc (VOC,HID) f32 -> (VP,HID) f16, zero rows beyond VOC
__global__ void k_cast_pad_fc(const float* __restrict__ src, _Float16* __restrict__ dst) {
    int i = blockIdx.x * blockDim.x + threadIdx.x;
    if (i < VP * HID) dst[i] = (i < VOC * HID) ? (_Float16)src[i] : (_Float16)0.f;
}

__global__ void k_zero_h(_Float16* __restrict__ a, _Float16* __restrict__ b) {
    int i = blockIdx.x * blockDim.x + threadIdx.x;
    if (i < BAT * HID) { a[i] = (_Float16)0.f; b[i] = (_Float16)0.f; }
}

// ---- layer-1 recurrent step:  h' = tanh(Wih1[:,x] + h@Whh1^T + bih + bhh) --
// grid 32 blocks x 256 thr = 256 waves; wave -> one 16x16 tile (4 mt x 64 nt)
__global__ void k_rnn1(const _Float16* __restrict__ hin, _Float16* __restrict__ hout,
                       const _Float16* __restrict__ Whh, const int* __restrict__ xs,
                       const float* __restrict__ Wih1, const float* __restrict__ bih,
                       const float* __restrict__ bhh, _Float16* __restrict__ y,
                       float* __restrict__ state) {
    const int wid = blockIdx.x * (blockDim.x >> 5) + (threadIdx.x >> 5);
    const int m0 = (wid & 3) * 16;
    const int n0 = (wid >> 2) * 16;
    v8f acc = {};
    const _Float16* arow = hin + (size_t)m0 * HID;
    const _Float16* wrow = Whh + (size_t)n0 * HID;
    #pragma unroll 4
    for (int k = 0; k < HID; k += 32) {
        v16h a = load_a16x32(arow + k, HID);
        v16h b = load_b32x16(wrow + k, HID);
        acc = wmma_f16(a, b, acc);
    }
    const int lane = threadIdx.x & 31;
    const int col  = n0 + (lane & 15);
    const int rb   = m0 + ((lane >> 4) << 3);
    const float bsum = bih[col] + bhh[col];
    #pragma unroll
    for (int i = 0; i < 8; ++i) {
        const int row = rb + i;                       // batch index
        float v = acc[i] + Wih1[(size_t)col * VOC + xs[row]] + bsum;
        float t = tanhf(v);
        _Float16 th = (_Float16)t;
        hout[(size_t)row * HID + col] = th;
        y[(size_t)row * HID + col]    = th;
        if (state) state[(size_t)row * HID + col] = t;
    }
}

// ---- layer-2 recurrent step: h' = tanh(y1@Wih2^T + h@Whh2^T + bih + bhh) ---
__global__ void k_rnn2(const _Float16* __restrict__ hin, _Float16* __restrict__ hout,
                       const _Float16* __restrict__ Whh, const _Float16* __restrict__ y1s,
                       const _Float16* __restrict__ Wih, const float* __restrict__ bih,
                       const float* __restrict__ bhh, _Float16* __restrict__ y2s,
                       float* __restrict__ state) {
    const int wid = blockIdx.x * (blockDim.x >> 5) + (threadIdx.x >> 5);
    const int m0 = (wid & 3) * 16;
    const int n0 = (wid >> 2) * 16;
    v8f acc = {};
    const _Float16* ah = hin + (size_t)m0 * HID;
    const _Float16* ax = y1s + (size_t)m0 * HID;
    const _Float16* wh = Whh + (size_t)n0 * HID;
    const _Float16* wx = Wih + (size_t)n0 * HID;
    #pragma unroll 4
    for (int k = 0; k < HID; k += 32) {
        v16h a = load_a16x32(ah + k, HID);
        v16h b = load_b32x16(wh + k, HID);
        acc = wmma_f16(a, b, acc);
    }
    #pragma unroll 4
    for (int k = 0; k < HID; k += 32) {
        v16h a = load_a16x32(ax + k, HID);
        v16h b = load_b32x16(wx + k, HID);
        acc = wmma_f16(a, b, acc);
    }
    const int lane = threadIdx.x & 31;
    const int col  = n0 + (lane & 15);
    const int rb   = m0 + ((lane >> 4) << 3);
    const float bsum = bih[col] + bhh[col];
    #pragma unroll
    for (int i = 0; i < 8; ++i) {
        const int row = rb + i;
        float t = tanhf(acc[i] + bsum);
        _Float16 th = (_Float16)t;
        hout[(size_t)row * HID + col] = th;
        y2s[(size_t)row * HID + col]  = th;
        if (state) state[(size_t)row * HID + col] = t;
    }
}

// ---- final projection: ans = Y2 @ Wfc^T + b_fc  (8192 x 1024 x 5000) -------
// Block = 8 waves = 128 rows x 64-col strip. B strip staged in LDS per
// K-chunk of 64 via async global->LDS copies; all 8 waves share it.
static constexpr int MT_FC = (SEQL * BAT) / 16;  // 512 m-tiles
static constexpr int ST_FC = VP / 64;            // 79 strips of 4 n-tiles
static constexpr int KC    = 64;                 // K chunk staged in LDS
static constexpr int BSTR  = KC + 8;             // padded row stride (halfs): 144B
                                                 // -> conflict-free b128 frag reads

// B-fragment read from LDS-staged strip. base = &Bs[nbase*BSTR + kk].
__device__ __forceinline__ v16h load_b_lds(const _Float16* base) {
    const int lane = threadIdx.x & 31;
    const int n  = lane & 15;
    const int hh = lane >> 4;
    const _Float16* q = base + n * BSTR + hh * 16;
    v8h lo = *reinterpret_cast<const v8h*>(q);
    v8h hi = *reinterpret_cast<const v8h*>(q + 8);
    return __builtin_shufflevector(lo, hi, 0,1,2,3,4,5,6,7,8,9,10,11,12,13,14,15);
}

__global__ void __launch_bounds__(256) k_fc(const _Float16* __restrict__ Y,
                                            const _Float16* __restrict__ Wt,
                                            const float* __restrict__ bias,
                                            float* __restrict__ out) {
    __shared__ __align__(16) _Float16 Bs[64 * BSTR];   // 9216 B

    const int tid  = threadIdx.x;
    const int wave = tid >> 5;
    const int lane = tid & 31;
    const int bm   = blockIdx.x % (MT_FC / 8);   // 64 m-blocks
    const int st   = blockIdx.x / (MT_FC / 8);   // 79 strips
    const int m0   = bm * 128 + wave * 16;
    const int n0   = st * 64;

    // staging map: each thread copies two 16B pieces per K-chunk
    const int lrow0 = tid >> 3;          // 0..31
    const int lrow1 = 32 + lrow0;        // 32..63
    const int lpart = (tid & 7) * 8;     // k offset in halfs (0..56)

    v8f acc0 = {}, acc1 = {}, acc2 = {}, acc3 = {};
    const _Float16* yb = Y  + (size_t)m0 * HID;
    const _Float16* wb = Wt + (size_t)n0 * HID;

    for (int kc = 0; kc < HID; kc += KC) {
        // stage B strip chunk: 64 rows x 64 halfs
        const _Float16* gsrc = wb + kc;
        copy16_g2lds(gsrc + (size_t)lrow0 * HID + lpart, &Bs[lrow0 * BSTR + lpart]);
        copy16_g2lds(gsrc + (size_t)lrow1 * HID + lpart, &Bs[lrow1 * BSTR + lpart]);
        wait_async_copies();
        __syncthreads();

        #pragma unroll
        for (int kk = 0; kk < KC; kk += 32) {
            v16h a  = load_a16x32(yb + kc + kk, HID);
            v16h b0 = load_b_lds(&Bs[ 0 * BSTR + kk]);
            v16h b1 = load_b_lds(&Bs[16 * BSTR + kk]);
            v16h b2 = load_b_lds(&Bs[32 * BSTR + kk]);
            v16h b3 = load_b_lds(&Bs[48 * BSTR + kk]);
            acc0 = wmma_f16(a, b0, acc0);
            acc1 = wmma_f16(a, b1, acc1);
            acc2 = wmma_f16(a, b2, acc2);
            acc3 = wmma_f16(a, b3, acc3);
        }
        __syncthreads();   // before next chunk overwrites Bs
    }

    const int cn = lane & 15;
    const int rb = m0 + ((lane >> 4) << 3);
    v8f accs[4] = {acc0, acc1, acc2, acc3};
    #pragma unroll
    for (int j = 0; j < 4; ++j) {
        const int col = n0 + j * 16 + cn;
        if (col >= VOC) continue;
        const float bv = bias[col];
        #pragma unroll
        for (int i = 0; i < 8; ++i) {
            const int row = rb + i;
            out[(size_t)row * VOC + col] = accs[j][i] + bv;
        }
    }
}

// ---------------------------------------------------------------------------
extern "C" void kernel_launch(void* const* d_in, const int* in_sizes, int n_in,
                              void* d_out, int out_size, void* d_ws, size_t ws_size,
                              hipStream_t stream) {
    const int*   x    = (const int*)  d_in[0];
    const float* Wih1 = (const float*)d_in[1];   // (HID, VOC)
    const float* Whh1 = (const float*)d_in[2];   // (HID, HID)
    const float* bih1 = (const float*)d_in[3];
    const float* bhh1 = (const float*)d_in[4];
    const float* Wih2 = (const float*)d_in[5];   // (HID, HID)
    const float* Whh2 = (const float*)d_in[6];   // (HID, HID)
    const float* bih2 = (const float*)d_in[7];
    const float* bhh2 = (const float*)d_in[8];
    const float* Wfc  = (const float*)d_in[9];   // (VOC, HID)
    const float* bfc  = (const float*)d_in[10];
    float* out = (float*)d_out;
    const size_t ANS = (size_t)SEQL * BAT * VOC;

    // workspace layout (256B aligned slices)
    char* ws = (char*)d_ws;
    size_t off = 0;
    auto take = [&](size_t bytes) -> char* {
        char* p = ws + off;
        off = (off + bytes + 255) & ~(size_t)255;
        return p;
    };
    _Float16* Whh1h = (_Float16*)take((size_t)HID * HID * 2);
    _Float16* Whh2h = (_Float16*)take((size_t)HID * HID * 2);
    _Float16* Wih2h = (_Float16*)take((size_t)HID * HID * 2);
    _Float16* Wfch  = (_Float16*)take((size_t)VP  * HID * 2);
    _Float16* h1a   = (_Float16*)take((size_t)BAT * HID * 2);
    _Float16* h1b   = (_Float16*)take((size_t)BAT * HID * 2);
    _Float16* h2a   = (_Float16*)take((size_t)BAT * HID * 2);
    _Float16* h2b   = (_Float16*)take((size_t)BAT * HID * 2);
    _Float16* y1h   = (_Float16*)take((size_t)SEQL * BAT * HID * 2);
    _Float16* y2h   = (_Float16*)take((size_t)SEQL * BAT * HID * 2);

    // weight casts + h0 = 0
    {
        int n = HID * HID;
        k_cast_f16<<<(n + 255) / 256, 256, 0, stream>>>(Whh1, Whh1h, n);
        k_cast_f16<<<(n + 255) / 256, 256, 0, stream>>>(Whh2, Whh2h, n);
        k_cast_f16<<<(n + 255) / 256, 256, 0, stream>>>(Wih2, Wih2h, n);
        int np = VP * HID;
        k_cast_pad_fc<<<(np + 255) / 256, 256, 0, stream>>>(Wfc, Wfch);
        int nh = BAT * HID;
        k_zero_h<<<(nh + 255) / 256, 256, 0, stream>>>(h1a, h2a);
    }

    // layer 1: 128 sequential steps, double-buffered h
    for (int s = 0; s < SEQL; ++s) {
        const _Float16* hin = (s & 1) ? h1b : h1a;
        _Float16*       hob = (s & 1) ? h1a : h1b;
        float* st = (s == SEQL - 1) ? (out + ANS) : nullptr;   // state[0]
        k_rnn1<<<32, 256, 0, stream>>>(hin, hob, Whh1h, x + (size_t)s * BAT,
                                       Wih1, bih1, bhh1,
                                       y1h + (size_t)s * BAT * HID, st);
    }

    // layer 2: 128 sequential steps (folds y1 @ Wih2^T into the K loop)
    for (int s = 0; s < SEQL; ++s) {
        const _Float16* hin = (s & 1) ? h2b : h2a;
        _Float16*       hob = (s & 1) ? h2a : h2b;
        float* st = (s == SEQL - 1) ? (out + ANS + (size_t)BAT * HID) : nullptr; // state[1]
        k_rnn2<<<32, 256, 0, stream>>>(hin, hob, Whh2h,
                                       y1h + (size_t)s * BAT * HID, Wih2h,
                                       bih2, bhh2,
                                       y2h + (size_t)s * BAT * HID, st);
    }

    // final projection (dominant GEMM): 64 m-blocks x 79 strips, 8 waves/block
    {
        const int blocks = (MT_FC / 8) * ST_FC;   // 5056
        k_fc<<<blocks, 256, 0, stream>>>(y2h, Wfch, bfc, out);
    }
}
